// NaiveSHA_50414326121154
// MI455X (gfx1250) — matrix operand order repaired and checked by reading, not compiled, not run on hardware
//
#include <hip/hip_runtime.h>
#include <hip/hip_bf16.h>

// ---------- CDNA5 WMMA types ----------
typedef __attribute__((ext_vector_type(16))) __bf16 v16bf;
typedef __attribute__((ext_vector_type(8)))  float  v8f;

union V16 {
    v16bf v;
    unsigned int u[8];
    unsigned short s[16];
    __bf16 h[16];
};

__device__ __forceinline__ unsigned short f2bf(float f) {
    __bf16 h = (__bf16)f;                       // native v_cvt_pk_bf16_f32 path
    return __builtin_bit_cast(unsigned short, h);
}

__device__ __forceinline__ v8f wmma_bf16(const V16& a, const V16& b, v8f c) {
    // v_wmma_f32_16x16x32_bf16: D = A(16x32) * B(32x16) + C(16x16 f32)
    return __builtin_amdgcn_wmma_f32_16x16x32_bf16(
        false, a.v, false, b.v, (short)0, c, false, false);
}

// ---------- CDNA5 async global->LDS copy (ASYNCcnt path) ----------
__device__ __forceinline__ void async_ld_b128(unsigned lds_off, const void* gaddr) {
    asm volatile("global_load_async_to_lds_b128 %0, %1, off"
                 :: "v"(lds_off), "v"(gaddr) : "memory");
}
__device__ __forceinline__ void wait_async0() {
    asm volatile("s_wait_asynccnt 0x0" ::: "memory");
}

// load a 32x16 B-operand tile (k-contiguous layout) from a bf16 row block
__device__ __forceinline__ void loadB16(V16& dst, const unsigned short* basep) {
    const unsigned int* wp = (const unsigned int*)basep;
    #pragma unroll
    for (int d = 0; d < 8; ++d) dst.u[d] = wp[(d & 3) + (d >> 2) * 8];
}

// Sizes for this problem
#define BATCH 4
#define SEQ   4096
#define DMODEL 512
#define DHEAD 64
#define NROWS (BATCH * SEQ)   // 16384

// ---------- Kernel 0: convert weights to bf16, k-contiguous (B-operand) layouts ----------
// WT  : [3][64][512]  (n-major, k contiguous)  for Wq,Wk,Wv  (W is [512][64] row-major)
// WoT : [512][64]     (n-major, k contiguous)  for Wo        (Wo is [64][512] row-major)
__global__ void convw_kernel(const float* __restrict__ Wq, const float* __restrict__ Wk,
                             const float* __restrict__ Wv, const float* __restrict__ Wo,
                             unsigned short* __restrict__ WT,
                             unsigned short* __restrict__ WoT)
{
    int idx = blockIdx.x * 256 + threadIdx.x;   // 0..32767
    int m = blockIdx.y;                         // 0..3
    if (m < 3) {
        const float* W = (m == 0) ? Wq : (m == 1) ? Wk : Wv;
        int n = idx >> 9;        // 0..63
        int k = idx & 511;       // 0..511
        WT[m * 32768 + n * 512 + k] = f2bf(W[k * 64 + n]);
    } else {
        int n = idx >> 6;        // 0..511
        int k = idx & 63;        // 0..63
        WoT[n * 64 + k] = f2bf(Wo[k * 512 + n]);
    }
}

// ---------- Kernel 1: QKV projection ----------
// 128 threads = 4 waves, 64 rows per block. Weight slab for the current k-chunk
// (3 mats x 64 cols x 32 k = 12KB bf16) is staged into LDS with async copies,
// double-buffered so chunk c+1 streams in while chunk c feeds 12 WMMAs per wave.
// B-operand LDS reads are 2-deep software pipelined against the WMMAs.
__global__ void qkv_kernel(const float* __restrict__ x,
                           const unsigned short* __restrict__ WT,
                           const float* __restrict__ bq, const float* __restrict__ bk,
                           const float* __restrict__ bv,
                           unsigned short* __restrict__ Qb,
                           unsigned short* __restrict__ Kb,
                           unsigned short* __restrict__ Vb)
{
    __shared__ unsigned short Wl[2][3 * 64 * 32];   // 2 x 12KB, [m*64+n][32k]

    int tid  = threadIdx.x;
    int wave = tid >> 5;
    int ln   = tid & 31;
    int hi   = ln >> 4, lo = ln & 15;
    int rowbase = (blockIdx.x * 4 + wave) * 16;
    int row = rowbase + lo;

    // async-stage chunk c of all three weight matrices into buffer `buf`
    auto stage = [&](int buf, int c) {
        #pragma unroll
        for (int i = 0; i < 6; ++i) {                 // 768 x b128 over 128 threads
            int id = tid + 128 * i;                   // 0..767
            int r  = id >> 2;                         // 0..191 = m*64+n
            int p  = id & 3;                          // 8-element part of the 32-k slab
            unsigned dst = (unsigned)(size_t)&Wl[buf][r * 32 + p * 8];
            const void* src = WT + (size_t)r * 512 + 32 * c + p * 8;
            async_ld_b128(dst, src);
        }
    };

    v8f acc[3][4] = {};
    stage(0, 0);

    for (int c = 0; c < 16; ++c) {                    // k-chunks of 32 over D=512
        wait_async0();
        __syncthreads();                              // chunk c resident for all waves
        if (c + 1 < 16) stage((c + 1) & 1, c + 1);    // overlap next chunk's DMA
        int buf = c & 1;

        // A operand: row `row`, K = 32c + {0..7,16..23} (lanes 0-15) / {8..15,24..31}
        V16 a;
        const float* xp = x + (size_t)row * DMODEL + 32 * c;
        if (c + 1 < 16) __builtin_prefetch(xp + 32, 0, 0);
        #pragma unroll
        for (int i = 0; i < 8; ++i) a.h[i]     = (__bf16)xp[8 * hi + i];
        #pragma unroll
        for (int i = 0; i < 8; ++i) a.h[8 + i] = (__bf16)xp[16 + 8 * hi + i];

        // 12 WMMAs, B tiles pipelined 2 deep through LDS
        const unsigned short* wl = &Wl[buf][(lo)*32 + 8 * hi];
        V16 bm[2];
        loadB16(bm[0], wl);                            // tile (m=0,t=0)
        #pragma unroll
        for (int i = 0; i < 12; ++i) {
            int m = i >> 2, t = i & 3;
            if (i + 1 < 12) {
                int m1 = (i + 1) >> 2, t1 = (i + 1) & 3;
                loadB16(bm[(i + 1) & 1], wl + (m1 * 64 + t1 * 16) * 32);
            }
            acc[m][t] = wmma_bf16(a, bm[i & 1], acc[m][t]);
        }
    }

    const float* biases[3] = { bq, bk, bv };
    unsigned short* outs[3] = { Qb, Kb, Vb };
    #pragma unroll
    for (int m = 0; m < 3; ++m) {
        float scale = (m == 0) ? 0.125f : 1.0f;       // fold 1/sqrt(dk) into Q
        #pragma unroll
        for (int t = 0; t < 4; ++t) {
            float bias = biases[m][t * 16 + lo];
            #pragma unroll
            for (int vr = 0; vr < 8; ++vr) {
                float val = (acc[m][t][vr] + bias) * scale;
                outs[m][(size_t)(rowbase + vr + 8 * hi) * DHEAD + t * 16 + lo] = f2bf(val);
            }
        }
    }
}

// ---------- Kernel 2: causal flash attention ----------
// WG = 256 threads = 8 waves; wave w owns query rows qblk*128 + 16w .. +15.
// K tile (32x64) streams to LDS via async global->LDS DMA; V tile is transposed
// into LDS (64x32, padded stride) so it serves directly as a WMMA B operand.
__global__ void attn_kernel(const unsigned short* __restrict__ Qb,
                            const unsigned short* __restrict__ Kb,
                            const unsigned short* __restrict__ Vb,
                            unsigned short* __restrict__ Ob)
{
    __shared__ unsigned short Kl[32 * 64];        // [key][dk]
    __shared__ unsigned short Vt[64 * 34];        // [dv][key], stride 34 (pad)
    __shared__ unsigned short Pl[8 * 16 * 32];    // per-wave P tile [m][key]

    int b    = blockIdx.y;
    int qblk = blockIdx.x;                        // 0..31
    int tid  = threadIdx.x;
    int wave = tid >> 5, ln = tid & 31, hi = ln >> 4, lo = ln & 15;
    int qr0  = qblk * 128 + wave * 16;            // first query row in batch
    size_t base = (size_t)b * SEQ * DHEAD;

    // Load Q A-operands (2 k-chunks of 32 over dk=64), rows qr0+lo
    V16 aq[2];
    {
        const unsigned int* qp = (const unsigned int*)(Qb + base + (size_t)(qr0 + lo) * DHEAD);
        #pragma unroll
        for (int c = 0; c < 2; ++c) {
            #pragma unroll
            for (int d = 0; d < 4; ++d) aq[c].u[d]     = qp[c * 16 + 4 * hi + d];
            #pragma unroll
            for (int d = 0; d < 4; ++d) aq[c].u[4 + d] = qp[c * 16 + 8 + 4 * hi + d];
        }
    }

    float m_r[8], l_r[8];
    #pragma unroll
    for (int vr = 0; vr < 8; ++vr) { m_r[vr] = -1e30f; l_r[vr] = 0.0f; }
    v8f acc[4] = {};

    int nj = (qblk + 1) * 4;                      // key blocks of 32 covering causal range
    for (int jb = 0; jb < nj; ++jb) {
        int j0 = jb * 32;
        __syncthreads();                          // protect K/V LDS reuse (WAR)
        {   // cooperative staging: 256 threads x 16B
            int r = tid >> 3, g = tid & 7;        // r: key row 0..31, g: 8-col group
            // K: straight copy via async DMA (no VGPR round-trip)
            async_ld_b128((unsigned)(size_t)&Kl[r * 64 + g * 8],
                          Kb + base + (size_t)(j0 + r) * DHEAD + g * 8);
            // V: load + transpose into [dv][key]
            uint4 vv = *(const uint4*)(Vb + base + (size_t)(j0 + r) * DHEAD + g * 8);
            const unsigned short* v8 = (const unsigned short*)&vv;
            #pragma unroll
            for (int i = 0; i < 8; ++i) Vt[(g * 8 + i) * 34 + r] = v8[i];
        }
        wait_async0();
        __syncthreads();

        // S = Q * K^T : 16x32 scores = 2 n-tiles, accumulated over 2 k-chunks;
        // B tiles pipelined 2 deep through LDS
        v8f sc[2] = {};
        {
            const unsigned short* kl = &Kl[lo * 64 + 8 * hi];
            V16 kb[2];
            loadB16(kb[0], kl);                        // (t=0,c=0)
            #pragma unroll
            for (int i = 0; i < 4; ++i) {
                int t = i >> 1, cc = i & 1;
                if (i + 1 < 4) {
                    int t1 = (i + 1) >> 1, c1 = (i + 1) & 1;
                    loadB16(kb[(i + 1) & 1], kl + t1 * 16 * 64 + c1 * 32);
                }
                sc[t] = wmma_bf16(aq[cc], kb[i & 1], sc[t]);
            }
        }

        // causal mask + online softmax (rows live across 16-lane halves)
        #pragma unroll
        for (int vr = 0; vr < 8; ++vr) {
            int irow = qr0 + vr + 8 * hi;
            float s0 = sc[0][vr]; if (j0 + lo      > irow) s0 = -1e30f;
            float s1 = sc[1][vr]; if (j0 + 16 + lo > irow) s1 = -1e30f;
            float rm = fmaxf(s0, s1);
            #pragma unroll
            for (int msk = 1; msk <= 8; msk <<= 1) rm = fmaxf(rm, __shfl_xor(rm, msk, 32));
            float mn = fmaxf(m_r[vr], rm);
            float alpha = __expf(m_r[vr] - mn);
            float p0 = __expf(s0 - mn);
            float p1 = __expf(s1 - mn);
            float rs = p0 + p1;
            #pragma unroll
            for (int msk = 1; msk <= 8; msk <<= 1) rs += __shfl_xor(rs, msk, 32);
            l_r[vr] = l_r[vr] * alpha + rs;
            m_r[vr] = mn;
            #pragma unroll
            for (int t = 0; t < 4; ++t) acc[t][vr] *= alpha;
            // spill P (C-layout) to LDS for A-layout reload
            Pl[wave * 512 + (vr + 8 * hi) * 32 + lo]      = f2bf(p0);
            Pl[wave * 512 + (vr + 8 * hi) * 32 + 16 + lo] = f2bf(p1);
        }
        __syncthreads();   // also orders per-wave P store->load (LDS in-order per wave)

        // O += P(16x32) * V(32x64) : A from Pl, B from Vt (pipelined 2 deep)
        V16 ap;
        {
            const unsigned int* pp = (const unsigned int*)&Pl[wave * 512 + lo * 32];
            #pragma unroll
            for (int d = 0; d < 4; ++d) ap.u[d]     = pp[4 * hi + d];
            #pragma unroll
            for (int d = 0; d < 4; ++d) ap.u[4 + d] = pp[8 + 4 * hi + d];
        }
        {
            const unsigned short* vl = &Vt[lo * 34 + 8 * hi];
            V16 bv_[2];
            loadB16(bv_[0], vl);
            #pragma unroll
            for (int t = 0; t < 4; ++t) {
                if (t + 1 < 4) loadB16(bv_[(t + 1) & 1], vl + (t + 1) * 16 * 34);
                acc[t] = wmma_bf16(ap, bv_[t & 1], acc[t]);
            }
        }
    }

    // normalize and store O (bf16, row-major [N][64])
    #pragma unroll
    for (int t = 0; t < 4; ++t) {
        #pragma unroll
        for (int vr = 0; vr < 8; ++vr) {
            float o = acc[t][vr] * (1.0f / l_r[vr]);
            Ob[base + (size_t)(qr0 + vr + 8 * hi) * DHEAD + t * 16 + lo] = f2bf(o);
        }
    }
}

// ---------- Kernel 3: output projection  out = O @ Wo + bo ----------
__global__ void oproj_kernel(const unsigned short* __restrict__ Ob,
                             const unsigned short* __restrict__ WoT,
                             const float* __restrict__ bo,
                             float* __restrict__ out)
{
    int wave = threadIdx.x >> 5, ln = threadIdx.x & 31, hi = ln >> 4, lo = ln & 15;
    int rowbase = (blockIdx.x * 4 + wave) * 16;
    int colbase = blockIdx.y * 64;

    V16 a[2];
    {
        const unsigned int* op = (const unsigned int*)(Ob + (size_t)(rowbase + lo) * DHEAD);
        #pragma unroll
        for (int c = 0; c < 2; ++c) {
            #pragma unroll
            for (int d = 0; d < 4; ++d) a[c].u[d]     = op[c * 16 + 4 * hi + d];
            #pragma unroll
            for (int d = 0; d < 4; ++d) a[c].u[4 + d] = op[c * 16 + 8 + 4 * hi + d];
        }
    }

    v8f acc[4] = {};
    {
        const unsigned short* wb = WoT + (size_t)(colbase + lo) * DHEAD + 8 * hi;
        V16 bm[2];
        loadB16(bm[0], wb);                            // (c=0,t=0)
        #pragma unroll
        for (int i = 0; i < 8; ++i) {
            int cc = i >> 2, t = i & 3;
            if (i + 1 < 8) {
                int c1 = (i + 1) >> 2, t1 = (i + 1) & 3;
                loadB16(bm[(i + 1) & 1], wb + (size_t)t1 * 16 * DHEAD + c1 * 32);
            }
            acc[t] = wmma_bf16(a[cc], bm[i & 1], acc[t]);
        }
    }

    #pragma unroll
    for (int t = 0; t < 4; ++t) {
        float bias = bo[colbase + t * 16 + lo];
        #pragma unroll
        for (int vr = 0; vr < 8; ++vr) {
            out[(size_t)(rowbase + vr + 8 * hi) * DMODEL + colbase + t * 16 + lo] =
                acc[t][vr] + bias;
        }
    }
}

// ---------- host launcher ----------
extern "C" void kernel_launch(void* const* d_in, const int* in_sizes, int n_in,
                              void* d_out, int out_size, void* d_ws, size_t ws_size,
                              hipStream_t stream) {
    (void)in_sizes; (void)n_in; (void)out_size; (void)ws_size;
    const float* x  = (const float*)d_in[0];
    const float* Wq = (const float*)d_in[1];
    const float* bq = (const float*)d_in[2];
    const float* Wk = (const float*)d_in[3];
    const float* bk = (const float*)d_in[4];
    const float* Wv = (const float*)d_in[5];
    const float* bv = (const float*)d_in[6];
    const float* Wo = (const float*)d_in[7];
    const float* bo = (const float*)d_in[8];
    float* out = (float*)d_out;

    // workspace layout (bytes)
    char* ws = (char*)d_ws;
    unsigned short* WT  = (unsigned short*)(ws);                       // 3*64*512*2   = 196608
    unsigned short* WoT = (unsigned short*)(ws + 196608);              // 512*64*2     = 65536
    unsigned short* Qb  = (unsigned short*)(ws + 262144);              // 16384*64*2   = 2097152
    unsigned short* Kb  = (unsigned short*)(ws + 262144 + 2097152);
    unsigned short* Vb  = (unsigned short*)(ws + 262144 + 2 * 2097152);
    unsigned short* Ob  = (unsigned short*)(ws + 262144 + 3 * 2097152);
    // total = 8,650,752 bytes

    convw_kernel<<<dim3(128, 4), 256, 0, stream>>>(Wq, Wk, Wv, Wo, WT, WoT);
    qkv_kernel  <<<dim3(256),    128, 0, stream>>>(x, WT, bq, bk, bv, Qb, Kb, Vb);
    attn_kernel <<<dim3(32, 4),  256, 0, stream>>>(Qb, Kb, Vb, Ob);
    oproj_kernel<<<dim3(256, 8), 128, 0, stream>>>(Ob, WoT, bo, out);
}
